// SNNVPR_8830452761075
// MI455X (gfx1250) — matrix operand order repaired and checked by reading, not compiled, not run on hardware
//
#include <hip/hip_runtime.h>
#include <math.h>
#include <stdint.h>

typedef __bf16 bf16;
typedef __attribute__((ext_vector_type(16))) __bf16 v16bf;
typedef __attribute__((ext_vector_type(8)))  float  v8f;

// ---------------- problem constants ----------------
constexpr int BB   = 64;        // batch
constexpr int CC   = 1024;      // channels
constexpr int NN   = 400;       // tokens / LIF time steps
constexpr int NP   = 416;       // 400 padded to multiple of 32 (13 k-chunks)
constexpr int NPAD = 448;       // 400 padded to 7*64 row tiles (no in-loop guards)
constexpr int MR   = BB * CC;   // 65536 flattened (b,c) rows
constexpr int YLD  = BB * NP;   // 26624, row stride of y/s2 per time step

// ---------------- WMMA helpers ----------------
__device__ __forceinline__ v8f vzero8() {
    v8f z;
#pragma unroll
    for (int i = 0; i < 8; ++i) z[i] = 0.f;
    return z;
}

__device__ __forceinline__ v8f wmma_bf16(v16bf a, v16bf b, v8f c) {
    return __builtin_amdgcn_wmma_f32_16x16x32_bf16(false, a, false, b, (short)0, c, false, false);
}

// A-fragment (16x32 bf16), per ISA 7.12.2: lanes 0-15 M=lane K={0..7,16..23},
// lanes 16-31 M=lane-16 K={8..15,24..31}. LDS tile row-major, stride 40 bf16 (80B: 16B aligned).
__device__ __forceinline__ v16bf frag_a(const bf16* t, int lane) {
    int m  = lane & 15;
    int kb = (lane >> 4) << 3;              // 0 or 8
    const bf16* p = t + m * 40;
    union { v16bf v; uint4 q[2]; } u;
    u.q[0] = *(const uint4*)(p + kb);       // K = kb .. kb+7
    u.q[1] = *(const uint4*)(p + 16 + kb);  // K = 16+kb .. 23+kb
    return u.v;
}

// B-fragment (32x16 bf16): lanes 0-15 col=lane K={0..7,16..23}; lanes 16-31 col=lane-16 K={8..15,24..31}.
__device__ __forceinline__ v16bf frag_b(const bf16* t, int lane) {
    int n  = lane & 15;
    int kb = (lane >> 4) << 3;              // 0 or 8
    const bf16* p = t + n * 40;
    union { v16bf v; uint4 q[2]; } u;
    u.q[0] = *(const uint4*)(p + kb);
    u.q[1] = *(const uint4*)(p + 16 + kb);
    return u.v;
}

// ---------------- CDNA5 async memory -> LDS (ASYNCcnt-tracked DMA) ----------------
__device__ __forceinline__ void async_ld16(unsigned int lds_off, const void* gaddr) {
    asm volatile("global_load_async_to_lds_b128 %0, %1, off"
                 :: "v"(lds_off), "v"(gaddr) : "memory");
}
__device__ __forceinline__ void s_wait_async3() { asm volatile("s_wait_asynccnt 0x3" ::: "memory"); }
__device__ __forceinline__ void s_wait_async2() { asm volatile("s_wait_asynccnt 0x2" ::: "memory"); }
__device__ __forceinline__ void s_wait_async0() { asm volatile("s_wait_asynccnt 0x0" ::: "memory"); }

__device__ __forceinline__ unsigned int lds_addr(const void* p) {
    return (unsigned int)(uintptr_t)p;      // generic LDS pointer: addr[31:0] is the LDS offset
}

// ---------------- elementwise / prep kernels ----------------

// f32 -> bf16 with zero padding
__global__ __launch_bounds__(256) void snn_cvt(const float* __restrict__ src, bf16* __restrict__ dst,
                                               int rows, int cols, int prows, int pcols) {
    int i = blockIdx.x * 256 + threadIdx.x;
    int total = prows * pcols;
    if (i >= total) return;
    int r = i / pcols, c = i - r * pcols;
    float v = (r < rows && c < cols) ? src[(size_t)r * cols + c] : 0.f;
    dst[i] = (bf16)v;
}

// fused BN params: y = acc*sc + of, where of = (conv_b - m)*sc + b
__global__ __launch_bounds__(256) void snn_bnprep(
    const float* __restrict__ c1b, const float* __restrict__ g1, const float* __restrict__ b1,
    const float* __restrict__ m1,  const float* __restrict__ v1,
    const float* __restrict__ c2b, const float* __restrict__ g2, const float* __restrict__ b2,
    const float* __restrict__ m2,  const float* __restrict__ v2,
    float* __restrict__ sc1, float* __restrict__ of1, float* __restrict__ sc2, float* __restrict__ of2) {
    int i = blockIdx.x * 256 + threadIdx.x;
    if (i < NP) {
        if (i < NN) {
            float s = g1[i] * rsqrtf(v1[i] + 1e-5f);
            sc1[i] = s; of1[i] = (c1b[i] - m1[i]) * s + b1[i];
        } else { sc1[i] = 0.f; of1[i] = 0.f; }
    }
    if (i < CC) {
        float s = g2[i] * rsqrtf(v2[i] + 1e-5f);
        sc2[i] = s; of2[i] = (c2b[i] - m2[i]) * s + b2[i];
    }
}

// LayerNorm over token dim per (b,c) row; writes bf16 padded to 416 (pad = 0)
__global__ __launch_bounds__(128) void snn_ln(const float* __restrict__ x,
                                              const float* __restrict__ g, const float* __restrict__ bta,
                                              bf16* __restrict__ H) {
    __shared__ float sh[10];
    int row = blockIdx.x;
    const float* xp = x + (size_t)row * NN;
    float s = 0.f, ss = 0.f;
    for (int j = threadIdx.x; j < NN; j += 128) { float v = xp[j]; s += v; ss += v * v; }
    for (int o = 16; o; o >>= 1) { s += __shfl_xor(s, o); ss += __shfl_xor(ss, o); }
    int wid = threadIdx.x >> 5, lane = threadIdx.x & 31;
    if (lane == 0) { sh[wid] = s; sh[4 + wid] = ss; }
    __syncthreads();
    if (threadIdx.x == 0) {
        float S = sh[0] + sh[1] + sh[2] + sh[3];
        float SS = sh[4] + sh[5] + sh[6] + sh[7];
        float mu = S * (1.f / NN);
        float var = SS * (1.f / NN) - mu * mu;
        sh[8] = mu; sh[9] = rsqrtf(var + 1e-5f);
    }
    __syncthreads();
    float mu = sh[8], ri = sh[9];
    bf16* hp = H + (size_t)row * NP;
    for (int j = threadIdx.x; j < NP; j += 128) {
        float h = (j < NN) ? ((xp[j] - mu) * ri * g[j] + bta[j]) : 0.f;
        hp[j] = (bf16)h;
    }
}

// LIF #1: per-(b,c) independent scan over N; spikes bf16 -> S[n][b][c]
__global__ __launch_bounds__(256) void snn_lif1(const float* __restrict__ x, bf16* __restrict__ S) {
    int t = blockIdx.x * 256 + threadIdx.x;        // t = b*1024 + c  (c fastest)
    const float* xp = x + (size_t)t * NN;
    float v = 0.f;
    for (int n = 0; n < NN; ++n) {
        v = 0.5f * (v + xp[n]);                    // v += (x - v)/tau, tau=2
        float sp = (v >= 1.f) ? 1.f : 0.f;         // H(v - 1)
        S[(size_t)n * MR + t] = (bf16)sp;
        v *= (1.f - sp);                           // hard reset
    }
}

// LIF #2: per-(b,o) scan on y[n][b][o]; writes s2[n][b][o] bf16, zeros for padded o
__global__ __launch_bounds__(256) void snn_lif2(const float* __restrict__ Y, bf16* __restrict__ S2) {
    int t = blockIdx.x * 256 + threadIdx.x;        // t = b*416 + o
    int o = t % NP;
    if (o >= NN) {
        for (int n = 0; n < NN; ++n) S2[(size_t)n * YLD + t] = (bf16)0.f;
        return;
    }
    float v = 0.f;
    for (int n = 0; n < NN; ++n) {
        v = 0.5f * (v + Y[(size_t)n * YLD + t]);
        float sp = (v >= 1.f) ? 1.f : 0.f;
        S2[(size_t)n * YLD + t] = (bf16)sp;
        v *= (1.f - sp);
    }
}

// ---------------- WMMA GEMM kernels (async double-buffered LDS, fully unrolled K) ----------------

// token-MLP GEMM: out[m,j] = A[m,:]·W[j,:]  (M=65536, J tiles over 448 padded rows, K=416)
// RELU=1: g1 = relu(.+b1) -> bf16[..][416] (pad cols 0)
// RELU=0: x1 = . + b2 + resid -> f32[..][400]
template <int RELU>
__global__ __launch_bounds__(256) void snn_mlp_gemm(
    const bf16* __restrict__ A, const bf16* __restrict__ W,
    const float* __restrict__ bias, const float* __restrict__ resid,
    bf16* __restrict__ outb, float* __restrict__ outf) {
    __shared__ bf16 lA[2][128 * 40];
    __shared__ bf16 lB[2][64 * 40];
    const int tid = threadIdx.x, lane = tid & 31, wid = tid >> 5;
    const int m0 = blockIdx.y * 128;
    const int j0 = blockIdx.x * 64;
    const int wm = (wid & 3) * 32, wj = (wid >> 2) * 32;
    v8f acc00 = vzero8(), acc01 = vzero8(), acc10 = vzero8(), acc11 = vzero8();

    // per-thread DMA slots: A rows r and r+64 (q-th 16B piece), B row r
    const int r = tid >> 2, q = tid & 3;
    const bf16* gA0 = A + (size_t)(m0 + r) * NP + q * 8;
    const bf16* gA1 = gA0 + (size_t)64 * NP;
    const bf16* gB0 = W + (size_t)(j0 + r) * NP + q * 8;   // W padded to 448 rows: no guard
    unsigned int la0[2], la1[2], lb0[2];
#pragma unroll
    for (int s = 0; s < 2; ++s) {
        la0[s] = lds_addr(&lA[s][r * 40 + q * 8]);
        la1[s] = lds_addr(&lA[s][(r + 64) * 40 + q * 8]);
        lb0[s] = lds_addr(&lB[s][r * 40 + q * 8]);
    }

    // prologue: tile 0 -> buffer 0
    async_ld16(la0[0], gA0);
    async_ld16(la1[0], gA1);
    async_ld16(lb0[0], gB0);

#pragma unroll
    for (int kt = 0; kt < 13; ++kt) {
        const int cur = kt & 1;            // constant after full unroll
        if (kt < 12) {
            const int k1 = (kt + 1) * 32;
            async_ld16(la0[cur ^ 1], gA0 + k1);
            async_ld16(la1[cur ^ 1], gA1 + k1);
            async_ld16(lb0[cur ^ 1], gB0 + k1);
            s_wait_async3();               // in-order completion: tile kt done
        } else {
            s_wait_async0();
        }
        __syncthreads();
        v16bf a0 = frag_a(&lA[cur][(wm + 0)  * 40], lane);
        v16bf a1 = frag_a(&lA[cur][(wm + 16) * 40], lane);
        v16bf b0 = frag_b(&lB[cur][(wj + 0)  * 40], lane);
        v16bf b1 = frag_b(&lB[cur][(wj + 16) * 40], lane);
        acc00 = wmma_bf16(a0, b0, acc00);
        acc01 = wmma_bf16(a0, b1, acc01);
        acc10 = wmma_bf16(a1, b0, acc10);
        acc11 = wmma_bf16(a1, b1, acc11);
        __syncthreads();
    }

    const int ml = (lane >> 4) * 8, nc = lane & 15;
#pragma unroll
    for (int mi = 0; mi < 2; ++mi) {
#pragma unroll
        for (int ji = 0; ji < 2; ++ji) {
            v8f acc = (mi == 0) ? (ji == 0 ? acc00 : acc01) : (ji == 0 ? acc10 : acc11);
            int gj = j0 + wj + ji * 16 + nc;
            size_t mbase = (size_t)m0 + wm + mi * 16 + ml;
#pragma unroll
            for (int v = 0; v < 8; ++v) {
                size_t gm = mbase + v;
                float val = acc[v];
                if (RELU) {
                    if (gj < NP) {
                        float o = (gj < NN) ? fmaxf(val + bias[gj], 0.f) : 0.f;
                        outb[gm * NP + gj] = (bf16)o;
                    }
                } else {
                    if (gj < NN) outf[gm * NN + gj] = val + bias[gj] + resid[gm * NN + gj];
                }
            }
        }
    }
}

// conv1 per time step: Y[n][b][o] = bn1( W1[o,:]·S[n][b,:] + conv1_b )   (K=1024)
__global__ __launch_bounds__(256) void snn_conv1(
    const bf16* __restrict__ W,            // [448][1024] padded bf16
    const bf16* __restrict__ S,            // [400][64][1024] spikes
    const float* __restrict__ sc, const float* __restrict__ ofs,
    float* __restrict__ Y) {               // [400][64][416]
    __shared__ bf16 lA[2][64 * 40];
    __shared__ bf16 lB[2][64 * 40];
    const int tid = threadIdx.x, lane = tid & 31, wid = tid >> 5;
    const int n = blockIdx.x;
    const int o0 = blockIdx.y * 64;
    const int wo = (wid & 3) * 16, wb = (wid >> 2) * 32;
    const bf16* Sn = S + (size_t)n * MR;
    v8f acc0 = vzero8(), acc1 = vzero8();

    const int r = tid >> 2, q = tid & 3;
    const bf16* gA = W + (size_t)(o0 + r) * CC + q * 8;
    const bf16* gB = Sn + (size_t)r * CC + q * 8;
    unsigned int la[2], lb[2];
#pragma unroll
    for (int s = 0; s < 2; ++s) {
        la[s] = lds_addr(&lA[s][r * 40 + q * 8]);
        lb[s] = lds_addr(&lB[s][r * 40 + q * 8]);
    }

    async_ld16(la[0], gA);
    async_ld16(lb[0], gB);

#pragma unroll
    for (int kt = 0; kt < 32; ++kt) {
        const int cur = kt & 1;
        if (kt < 31) {
            const int k1 = (kt + 1) * 32;
            async_ld16(la[cur ^ 1], gA + k1);
            async_ld16(lb[cur ^ 1], gB + k1);
            s_wait_async2();
        } else {
            s_wait_async0();
        }
        __syncthreads();
        v16bf a  = frag_a(&lA[cur][wo * 40], lane);
        v16bf b0 = frag_b(&lB[cur][(wb + 0)  * 40], lane);
        v16bf b1 = frag_b(&lB[cur][(wb + 16) * 40], lane);
        acc0 = wmma_bf16(a, b0, acc0);
        acc1 = wmma_bf16(a, b1, acc1);
        __syncthreads();
    }

    const int ml = (lane >> 4) * 8, nc = lane & 15;
#pragma unroll
    for (int ji = 0; ji < 2; ++ji) {
        v8f acc = ji ? acc1 : acc0;
        int b = wb + ji * 16 + nc;
#pragma unroll
        for (int v = 0; v < 8; ++v) {
            int o = o0 + wo + ml + v;
            if (o < NN) Y[(size_t)n * YLD + (size_t)b * NP + o] = acc[v] * sc[o] + ofs[o];
        }
    }
}

// conv2 per time step + BN + residual add: X1[b][c][n] += bn2( W2[c,:]·S2[n][b,:] + conv2_b )
__global__ __launch_bounds__(256) void snn_conv2(
    const bf16* __restrict__ W,            // [1024][416] bf16
    const bf16* __restrict__ S2,           // [400][64][416] spikes (pad o zeroed)
    const float* __restrict__ sc, const float* __restrict__ ofs,
    float* __restrict__ X1) {
    __shared__ bf16 lA[2][64 * 40];
    __shared__ bf16 lB[2][64 * 40];
    const int tid = threadIdx.x, lane = tid & 31, wid = tid >> 5;
    const int n = blockIdx.x;
    const int c0 = blockIdx.y * 64;
    const int wo = (wid & 3) * 16, wb = (wid >> 2) * 32;
    const bf16* Sn = S2 + (size_t)n * YLD;
    v8f acc0 = vzero8(), acc1 = vzero8();

    const int r = tid >> 2, q = tid & 3;
    const bf16* gA = W + (size_t)(c0 + r) * NP + q * 8;
    const bf16* gB = Sn + (size_t)r * NP + q * 8;
    unsigned int la[2], lb[2];
#pragma unroll
    for (int s = 0; s < 2; ++s) {
        la[s] = lds_addr(&lA[s][r * 40 + q * 8]);
        lb[s] = lds_addr(&lB[s][r * 40 + q * 8]);
    }

    async_ld16(la[0], gA);
    async_ld16(lb[0], gB);

#pragma unroll
    for (int kt = 0; kt < 13; ++kt) {
        const int cur = kt & 1;
        if (kt < 12) {
            const int k1 = (kt + 1) * 32;
            async_ld16(la[cur ^ 1], gA + k1);
            async_ld16(lb[cur ^ 1], gB + k1);
            s_wait_async2();
        } else {
            s_wait_async0();
        }
        __syncthreads();
        v16bf a  = frag_a(&lA[cur][wo * 40], lane);
        v16bf b0 = frag_b(&lB[cur][(wb + 0)  * 40], lane);
        v16bf b1 = frag_b(&lB[cur][(wb + 16) * 40], lane);
        acc0 = wmma_bf16(a, b0, acc0);
        acc1 = wmma_bf16(a, b1, acc1);
        __syncthreads();
    }

    const int ml = (lane >> 4) * 8, nc = lane & 15;
#pragma unroll
    for (int ji = 0; ji < 2; ++ji) {
        v8f acc = ji ? acc1 : acc0;
        int b = wb + ji * 16 + nc;
#pragma unroll
        for (int v = 0; v < 8; ++v) {
            int c = c0 + wo + ml + v;
            size_t idx = ((size_t)b * CC + c) * NN + n;
            X1[idx] += acc[v] * sc[c] + ofs[c];
        }
    }
}

// ---------------- projection tail ----------------

// xr[b,c,r] = sum_n x1[b,c,n] * rp_w[r,n]   (row_proj first: shrinks N of channel_proj to 4)
__global__ __launch_bounds__(256) void snn_rowproj(const float* __restrict__ X1,
                                                   const float* __restrict__ rpw,
                                                   float* __restrict__ xr) {
    int t = blockIdx.x * 256 + threadIdx.x;        // t = b*1024 + c
    const float* xp = X1 + (size_t)t * NN;
    float a0 = 0.f, a1 = 0.f, a2 = 0.f, a3 = 0.f;
    for (int n = 0; n < NN; ++n) {
        float xv = xp[n];
        a0 += xv * rpw[n];
        a1 += xv * rpw[NN + n];
        a2 += xv * rpw[2 * NN + n];
        a3 += xv * rpw[3 * NN + n];
    }
    float* o = xr + (size_t)t * 4;
    o[0] = a0; o[1] = a1; o[2] = a2; o[3] = a3;
}

// out[b,o,r] = sum_c cp_w[o,c]*xr[b,c,r] + cp_b[o]*sum_n rp_w[r,n] + rp_b[r]; L2-normalize per b
__global__ __launch_bounds__(256) void snn_final(const float* __restrict__ xr,
                                                 const float* __restrict__ cpw, const float* __restrict__ cpb,
                                                 const float* __restrict__ rpw, const float* __restrict__ rpb,
                                                 float* __restrict__ out) {
    __shared__ float xl[4096];
    __shared__ float swr[4];
    __shared__ float red[9];
    int b = blockIdx.x, tid = threadIdx.x;
    const float* xb = xr + (size_t)b * 4096;
    for (int i = tid; i < 1024; i += 256)
        *(float4*)&xl[i * 4] = *(const float4*)&xb[i * 4];
    if (tid < 4) {
        float s = 0.f;
        for (int n = 0; n < NN; ++n) s += rpw[tid * NN + n];
        swr[tid] = s;
    }
    __syncthreads();

    float vals[8];
    float ssq = 0.f;
#pragma unroll
    for (int k = 0; k < 8; ++k) {
        int i = tid * 8 + k;
        int o = i >> 2, r = i & 3;
        const float* wrow = cpw + (size_t)o * CC;
        float acc = 0.f;
        for (int c = 0; c < CC; ++c) acc += wrow[c] * xl[c * 4 + r];
        acc += cpb[o] * swr[r] + rpb[r];
        vals[k] = acc;
        ssq += acc * acc;
    }
    for (int o = 16; o; o >>= 1) ssq += __shfl_xor(ssq, o);
    int wid = tid >> 5, lane = tid & 31;
    if (lane == 0) red[wid] = ssq;
    __syncthreads();
    if (tid == 0) {
        float t = 0.f;
        for (int i = 0; i < 8; ++i) t += red[i];
        red[8] = 1.f / fmaxf(sqrtf(t), 1e-12f);
    }
    __syncthreads();
    float scale = red[8];
#pragma unroll
    for (int k = 0; k < 8; ++k) out[(size_t)b * 2048 + tid * 8 + k] = vals[k] * scale;
}

// ---------------- workspace layout ----------------
constexpr size_t aln(size_t v) { return (v + 255) & ~(size_t)255; }
constexpr size_t OFF_X1  = 0;                                   // f32 [65536][400]
constexpr size_t SZ_X1   = aln((size_t)MR * NN * 4);
constexpr size_t OFF_HB  = OFF_X1 + SZ_X1;                      // bf16 h [65536][416] -> reused as spikes S [400][64][1024]
constexpr size_t SZ_HB   = aln((size_t)MR * NP * 2);
constexpr size_t OFF_GB  = OFF_HB + SZ_HB;                      // bf16 g1 [65536][416] -> reused as Y f32 [400][64][416]
constexpr size_t SZ_GB   = SZ_HB;
constexpr size_t OFF_S2  = OFF_GB + SZ_GB;                      // bf16 s2 [400][64][416]
constexpr size_t SZ_S2   = aln((size_t)NN * YLD * 2);
constexpr size_t OFF_W1P = OFF_S2 + SZ_S2;                      // bf16 [448][416]
constexpr size_t SZ_WNN  = aln((size_t)NPAD * NP * 2);
constexpr size_t OFF_W2P = OFF_W1P + SZ_WNN;
constexpr size_t OFF_C1W = OFF_W2P + SZ_WNN;                    // bf16 [448][1024]
constexpr size_t SZ_C1W  = aln((size_t)NPAD * CC * 2);
constexpr size_t OFF_C2W = OFF_C1W + SZ_C1W;                    // bf16 [1024][416]
constexpr size_t SZ_C2W  = aln((size_t)CC * NP * 2);
constexpr size_t OFF_SC1 = OFF_C2W + SZ_C2W;                    // f32 [416]
constexpr size_t OFF_OF1 = OFF_SC1 + aln(NP * 4);
constexpr size_t OFF_SC2 = OFF_OF1 + aln(NP * 4);               // f32 [1024]
constexpr size_t OFF_OF2 = OFF_SC2 + aln(CC * 4);
constexpr size_t OFF_XR  = OFF_OF2 + aln(CC * 4);               // f32 [65536][4]

extern "C" void kernel_launch(void* const* d_in, const int* in_sizes, int n_in,
                              void* d_out, int out_size, void* d_ws, size_t ws_size,
                              hipStream_t stream) {
    const float* x    = (const float*)d_in[0];
    const float* ln_g = (const float*)d_in[1];
    const float* ln_b = (const float*)d_in[2];
    const float* w1   = (const float*)d_in[3];
    const float* b1   = (const float*)d_in[4];
    const float* w2   = (const float*)d_in[5];
    const float* b2   = (const float*)d_in[6];
    const float* c1w  = (const float*)d_in[7];
    const float* c1b  = (const float*)d_in[8];
    const float* bn1g = (const float*)d_in[9];
    const float* bn1b = (const float*)d_in[10];
    const float* bn1m = (const float*)d_in[11];
    const float* bn1v = (const float*)d_in[12];
    const float* c2w  = (const float*)d_in[13];
    const float* c2b  = (const float*)d_in[14];
    const float* bn2g = (const float*)d_in[15];
    const float* bn2b = (const float*)d_in[16];
    const float* bn2m = (const float*)d_in[17];
    const float* bn2v = (const float*)d_in[18];
    const float* cpw  = (const float*)d_in[19];
    const float* cpb  = (const float*)d_in[20];
    const float* rpw  = (const float*)d_in[21];
    const float* rpb  = (const float*)d_in[22];
    float* out = (float*)d_out;

    char* ws = (char*)d_ws;
    float* X1  = (float*)(ws + OFF_X1);
    bf16*  HB  = (bf16*) (ws + OFF_HB);   // h, then spikes S
    bf16*  GB  = (bf16*) (ws + OFF_GB);   // g1 (bf16) -> later Y (f32)
    float* YB  = (float*)(ws + OFF_GB);
    bf16*  S2  = (bf16*) (ws + OFF_S2);
    bf16*  W1P = (bf16*) (ws + OFF_W1P);
    bf16*  W2P = (bf16*) (ws + OFF_W2P);
    bf16*  C1W = (bf16*) (ws + OFF_C1W);
    bf16*  C2W = (bf16*) (ws + OFF_C2W);
    float* SC1 = (float*)(ws + OFF_SC1);
    float* OF1 = (float*)(ws + OFF_OF1);
    float* SC2 = (float*)(ws + OFF_SC2);
    float* OF2 = (float*)(ws + OFF_OF2);
    float* XR  = (float*)(ws + OFF_XR);

    // weight conversion (zero-padded so GEMM hot loops need no guards) + BN prep
    snn_cvt<<<(NPAD * NP + 255) / 256, 256, 0, stream>>>(w1, W1P, NN, NN, NPAD, NP);
    snn_cvt<<<(NPAD * NP + 255) / 256, 256, 0, stream>>>(w2, W2P, NN, NN, NPAD, NP);
    snn_cvt<<<(NPAD * CC + 255) / 256, 256, 0, stream>>>(c1w, C1W, NN, CC, NPAD, CC);
    snn_cvt<<<(CC * NP + 255) / 256, 256, 0, stream>>>(c2w, C2W, CC, NN, CC, NP);
    snn_bnprep<<<4, 256, 0, stream>>>(c1b, bn1g, bn1b, bn1m, bn1v,
                                      c2b, bn2g, bn2b, bn2m, bn2v, SC1, OF1, SC2, OF2);

    // token-mixing MLP (WMMA, async LDS streaming)
    snn_ln<<<MR, 128, 0, stream>>>(x, ln_g, ln_b, HB);
    snn_mlp_gemm<1><<<dim3(7, MR / 128), 256, 0, stream>>>(HB, W1P, b1, nullptr, GB, nullptr);
    snn_mlp_gemm<0><<<dim3(7, MR / 128), 256, 0, stream>>>(GB, W2P, b2, x, nullptr, X1);

    // spiking branch (LIF scans hoisted out of the batched WMMA GEMMs)
    snn_lif1<<<MR / 256, 256, 0, stream>>>(x, HB);                      // spikes -> HB
    snn_conv1<<<dim3(NN, 7), 256, 0, stream>>>(C1W, HB, SC1, OF1, YB);  // Y -> GB space
    snn_lif2<<<(BB * NP) / 256, 256, 0, stream>>>(YB, S2);
    snn_conv2<<<dim3(NN, CC / 64), 256, 0, stream>>>(C2W, S2, SC2, OF2, X1);

    // projection tail (row_proj before channel_proj: linear ops commute)
    snn_rowproj<<<MR / 256, 256, 0, stream>>>(X1, rpw, XR);
    snn_final<<<BB, 256, 0, stream>>>(XR, cpw, cpb, rpw, rpb, out);
}
